// EnhancedVectorQuantizer_25151328485489
// MI455X (gfx1250) — matrix operand order repaired and checked by reading, not compiled, not run on hardware
//
#include <hip/hip_runtime.h>
#include <hip/hip_bf16.h>
#include <stddef.h>
#include <stdint.h>

// ---------------------------------------------------------------------------
// Problem constants (from reference): z[4,2048,1024] f32, emb[8192,1024] f32
// ---------------------------------------------------------------------------
#define NROWS 8192      // 4*2048 flattened z rows
#define KCODES 8192     // number of codebook entries
#define DDIM 1024       // embedding dim (reduction)
#define TILE 128        // rows/cols per block tile
#define KT 32           // k-step (bf16 16x16x32)
#define NKS (DDIM / KT) // 32 k-steps
#define CSPLIT 8        // column splits (blockIdx.y)
#define PLANE 4096      // elems per LDS tile plane (128 rows x 32 bf16)

typedef __bf16 bf16_t;
typedef __attribute__((ext_vector_type(4)))  __bf16 v4bf;
typedef __attribute__((ext_vector_type(8)))  __bf16 v8bf;
typedef __attribute__((ext_vector_type(16))) __bf16 v16bf;
typedef __attribute__((ext_vector_type(8)))  float  v8f;
typedef __attribute__((ext_vector_type(4)))  int    v4i;

// ---- CDNA5 async global->LDS path (probe via __has_builtin, safe fallback) --
#if defined(__has_builtin)
#if __has_builtin(__builtin_amdgcn_global_load_async_to_lds_b128)
#define VQ_HAVE_ASYNC 1
#endif
#endif
#ifndef VQ_HAVE_ASYNC
#define VQ_HAVE_ASYNC 0
#endif

#if VQ_HAVE_ASYNC
#if __has_builtin(__builtin_amdgcn_s_wait_asynccnt)
#define VQ_WAIT_ASYNC() __builtin_amdgcn_s_wait_asynccnt(0)
#else
#define VQ_WAIT_ASYNC() asm volatile("s_wait_asynccnt 0x0" ::: "memory")
#endif
#else
#define VQ_WAIT_ASYNC() ((void)0)
#endif

static __device__ inline void async_cp16(const bf16_t* g, bf16_t* l) {
#if VQ_HAVE_ASYNC
  // builtin signature (per hipcc diagnostic): pointers to 16B int vectors
  __builtin_amdgcn_global_load_async_to_lds_b128((v4i*)g, (v4i*)l, 0, 0);
#else
  *(float4*)l = *(const float4*)g;   // sync fallback through VGPRs
#endif
}

static __device__ inline v16bf cat8(v8bf a, v8bf b) {
  return __builtin_shufflevector(a, b, 0,1,2,3,4,5,6,7,8,9,10,11,12,13,14,15);
}

// XOR swizzle of the 16B k-chunk within a row: kills LDS bank conflicts on
// both the per-lane async stores and the 16-lane fragment reads.
static __device__ inline int swz(int row, int koff) {
  return koff ^ (((row >> 2) & 3) << 3);
}

// ---------------------------------------------------------------------------
// Kernel 0: split fp32 -> bf16 hi + bf16 lo planes (done ONCE; removes all
// per-iteration VALU conversion from the GEMM hot loop).
// ---------------------------------------------------------------------------
__global__ __launch_bounds__(256) void vq_precvt(const float* __restrict__ src,
                                                 bf16_t* __restrict__ hi,
                                                 bf16_t* __restrict__ lo) {
  const size_t i = ((size_t)blockIdx.x * 256 + threadIdx.x) * 4;
  const float4 f = *(const float4*)(src + i);
  const float fp[4] = {f.x, f.y, f.z, f.w};
  v4bf h, l;
#pragma unroll
  for (int k = 0; k < 4; ++k) {
    const bf16_t hx = (bf16_t)fp[k];
    h[k] = hx;
    l[k] = (bf16_t)(fp[k] - (float)hx);
  }
  *(v4bf*)(hi + i) = h;
  *(v4bf*)(lo + i) = l;
}

// ---------------------------------------------------------------------------
// Kernel 1: e_norm[k] = sum(emb[k,:]^2) in fp32.
// ---------------------------------------------------------------------------
__global__ __launch_bounds__(256) void vq_enorm(const float* __restrict__ emb,
                                                float* __restrict__ enorm) {
  const int row = blockIdx.x;
  const float4 v = ((const float4*)(emb + (size_t)row * DDIM))[threadIdx.x];
  float s = v.x * v.x + v.y * v.y + v.z * v.z + v.w * v.w;
#pragma unroll
  for (int m = 16; m >= 1; m >>= 1) s += __shfl_xor(s, m, 32);
  __shared__ float red[8];
  if ((threadIdx.x & 31) == 0) red[threadIdx.x >> 5] = s;
  __syncthreads();
  if (threadIdx.x == 0) {
    float t = 0.f;
#pragma unroll
    for (int i = 0; i < 8; ++i) t += red[i];
    enorm[row] = t;
  }
}

// ---------------------------------------------------------------------------
// GEMM helpers
// ---------------------------------------------------------------------------
// Issue one 128x32 tile (4 planes: Ah Al Bh Bl) into an LDS buffer.
// 256 threads x 8 chunks of 16B = 8 KB/plane.
static __device__ inline void issue_tile(const bf16_t* __restrict__ zh,
                                         const bf16_t* __restrict__ zl,
                                         const bf16_t* __restrict__ eh,
                                         const bf16_t* __restrict__ el,
                                         bf16_t* __restrict__ sbuf,
                                         int tid, int rowBase, int colBase,
                                         int kOff) {
  const int r0 = tid >> 2;
  const int ko = (tid & 3) * 8;
#pragma unroll
  for (int i = 0; i < 2; ++i) {
    const int row = r0 + 64 * i;
    const size_t ga = (size_t)(rowBase + row) * DDIM + kOff + ko;
    const size_t gb = (size_t)(colBase + row) * DDIM + kOff + ko;
    const int ls = row * KT + swz(row, ko);
    async_cp16(zh + ga, sbuf + 0 * PLANE + ls);
    async_cp16(zl + ga, sbuf + 1 * PLANE + ls);
    async_cp16(eh + gb, sbuf + 2 * PLANE + ls);
    async_cp16(el + gb, sbuf + 3 * PLANE + ls);
  }
}

// 24 WMMAs on one LDS buffer: acc[r][c] += zh*eh + zh*el + zl*eh (bf16x3).
static __device__ inline void compute_tile(const bf16_t* __restrict__ sbuf,
                                           v8f (&acc)[2][4],
                                           int wrow, int wcol, int half,
                                           int l15) {
  v16bf ah[2], al[2];
#pragma unroll
  for (int r = 0; r < 2; ++r) {
    const int row = wrow * 32 + r * 16 + l15;   // A lane row (M = l15 group)
    const int s0 = row * KT + swz(row, half * 8);
    const int s1 = row * KT + swz(row, 16 + half * 8);
    ah[r] = cat8(*(const v8bf*)(sbuf + s0), *(const v8bf*)(sbuf + s1));
    al[r] = cat8(*(const v8bf*)(sbuf + PLANE + s0),
                 *(const v8bf*)(sbuf + PLANE + s1));
  }
  v16bf bh[4], bl[4];
#pragma unroll
  for (int c = 0; c < 4; ++c) {
    const int row = wcol * 64 + c * 16 + l15;   // B lane col, K in [half*8,+8) and [16+half*8,+8)
    const int s0 = row * KT + swz(row, half * 8);
    const int s1 = row * KT + swz(row, 16 + half * 8);
    bh[c] = cat8(*(const v8bf*)(sbuf + 2 * PLANE + s0),
                 *(const v8bf*)(sbuf + 2 * PLANE + s1));
    bl[c] = cat8(*(const v8bf*)(sbuf + 3 * PLANE + s0),
                 *(const v8bf*)(sbuf + 3 * PLANE + s1));
  }
#pragma unroll
  for (int r = 0; r < 2; ++r)
#pragma unroll
    for (int c = 0; c < 4; ++c) {
      acc[r][c] = __builtin_amdgcn_wmma_f32_16x16x32_bf16(
          false, ah[r], false, bh[c], (short)0, acc[r][c], false, false);
      acc[r][c] = __builtin_amdgcn_wmma_f32_16x16x32_bf16(
          false, ah[r], false, bl[c], (short)0, acc[r][c], false, false);
      acc[r][c] = __builtin_amdgcn_wmma_f32_16x16x32_bf16(
          false, al[r], false, bh[c], (short)0, acc[r][c], false, false);
    }
}

// ---------------------------------------------------------------------------
// Kernel 2: double-buffered async-fed WMMA GEMM + streaming argmin.
// Grid: (NROWS/TILE, CSPLIT), block 256 threads = 8 waves.
// Wave w: rows [(w>>1)*32,+32) x cols [(w&1)*64,+64) of a 128x128 tile.
// ---------------------------------------------------------------------------
__global__ __launch_bounds__(256) void vq_gemm_argmin(
    const bf16_t* __restrict__ zh, const bf16_t* __restrict__ zl,
    const bf16_t* __restrict__ eh, const bf16_t* __restrict__ el,
    const float* __restrict__ enorm,
    float* __restrict__ partS, int* __restrict__ partI) {
  __shared__ __align__(16) bf16_t smem[2 * 4 * PLANE];   // 64 KB, 2 buffers
  bf16_t* const buf0 = smem;
  bf16_t* const buf1 = smem + 4 * PLANE;

  const int tid  = threadIdx.x;
  const int lane = tid & 31;
  const int wid  = tid >> 5;
  const int half = lane >> 4;
  const int l15  = lane & 15;
  const int wrow = wid >> 1;       // 0..3
  const int wcol = wid & 1;        // 0..1

  const int rowBase = blockIdx.x * TILE;
  const int ct0     = blockIdx.y * (KCODES / TILE / CSPLIT);

  float best[2][8];
  int   bidx[2][8];
#pragma unroll
  for (int r = 0; r < 2; ++r)
#pragma unroll
    for (int j = 0; j < 8; ++j) { best[r][j] = 3.0e38f; bidx[r][j] = 0; }

  for (int ctl = 0; ctl < KCODES / TILE / CSPLIT; ++ctl) {
    const int colBase = (ct0 + ctl) * TILE;
    v8f acc[2][4];
#pragma unroll
    for (int r = 0; r < 2; ++r)
#pragma unroll
      for (int c = 0; c < 4; ++c)
#pragma unroll
        for (int j = 0; j < 8; ++j) acc[r][c][j] = 0.f;

    // prologue: stream tile 0 into buffer 0
    issue_tile(zh, zl, eh, el, buf0, tid, rowBase, colBase, 0);

#pragma unroll 1
    for (int ks2 = 0; ks2 < NKS / 2; ++ks2) {
      const int k0 = ks2 * 2;
      // ---- even step: compute buf0, stream k0+1 into buf1 ----
      VQ_WAIT_ASYNC();          // tile k0 landed (this wave's loads)
      __syncthreads();          // ... for every wave; buf1 readers also done
      issue_tile(zh, zl, eh, el, buf1, tid, rowBase, colBase, (k0 + 1) * KT);
      compute_tile(buf0, acc, wrow, wcol, half, l15);
      // ---- odd step: compute buf1, stream k0+2 into buf0 ----
      VQ_WAIT_ASYNC();
      __syncthreads();
      if (k0 + 2 < NKS)
        issue_tile(zh, zl, eh, el, buf0, tid, rowBase, colBase, (k0 + 2) * KT);
      compute_tile(buf1, acc, wrow, wcol, half, l15);
    }

    // streaming argmin: score = ||e||^2 - 2*dot  (||z||^2 is row-constant)
#pragma unroll
    for (int c = 0; c < 4; ++c) {
      const int col = colBase + wcol * 64 + c * 16 + l15;
      const float en = enorm[col];
#pragma unroll
      for (int r = 0; r < 2; ++r)
#pragma unroll
        for (int j = 0; j < 8; ++j) {
          const float score = fmaf(-2.f, acc[r][c][j], en);
          if (score < best[r][j]) { best[r][j] = score; bidx[r][j] = col; }
        }
    }
  }

  // merge scratch aliases buffer 0 (dead: last compute tile is always buf1)
  float* const redS = (float*)buf0;
  int*   const redI = (int*)(buf0 + PLANE);

#pragma unroll
  for (int r = 0; r < 2; ++r)
#pragma unroll
    for (int j = 0; j < 8; ++j) {
      float s = best[r][j];
      int  bi = bidx[r][j];
#pragma unroll
      for (int m = 1; m < 16; m <<= 1) {
        const float os = __shfl_xor(s, m, 32);
        const int   oi = __shfl_xor(bi, m, 32);
        if (os < s || (os == s && oi < bi)) { s = os; bi = oi; }
      }
      if (l15 == 0) {
        const int rl = wrow * 32 + r * 16 + j + 8 * half;   // local row 0..127
        redS[rl * 2 + wcol] = s;
        redI[rl * 2 + wcol] = bi;
      }
    }
  __syncthreads();
  if (tid < TILE) {
    float s0 = redS[tid * 2];
    int   i0 = redI[tid * 2];
    const float s1 = redS[tid * 2 + 1];
    const int   i1 = redI[tid * 2 + 1];
    if (s1 < s0 || (s1 == s0 && i1 < i0)) { s0 = s1; i0 = i1; }
    const int gr = rowBase + tid;
    partS[(size_t)gr * CSPLIT + blockIdx.y] = s0;
    partI[(size_t)gr * CSPLIT + blockIdx.y] = i0;
  }
}

// ---------------------------------------------------------------------------
// Kernel 3: fold the CSPLIT per-row candidates (first-min tie-break).
// ---------------------------------------------------------------------------
__global__ __launch_bounds__(256) void vq_merge(const float* __restrict__ partS,
                                                const int* __restrict__ partI,
                                                int* __restrict__ argmin) {
  const int row = blockIdx.x * blockDim.x + threadIdx.x;
  float s = partS[(size_t)row * CSPLIT];
  int  bi = partI[(size_t)row * CSPLIT];
#pragma unroll
  for (int k = 1; k < CSPLIT; ++k) {
    const float os = partS[(size_t)row * CSPLIT + k];
    const int   oi = partI[(size_t)row * CSPLIT + k];
    if (os < s || (os == s && oi < bi)) { s = os; bi = oi; }
  }
  argmin[row] = bi;
}

// ---------------------------------------------------------------------------
// Kernel 4: gather quantized rows, straight-through output z + (q - z),
// per-row squared-error partial (no atomics -> replay-deterministic).
// ---------------------------------------------------------------------------
__global__ __launch_bounds__(256) void vq_quantize(
    const float* __restrict__ z, const float* __restrict__ emb,
    const int* __restrict__ argmin, float* __restrict__ out,
    float* __restrict__ partials) {
  const int row  = blockIdx.x;
  const int code = argmin[row];
  const float4 e  = ((const float4*)(emb + (size_t)code * DDIM))[threadIdx.x];
  const float4 zz = ((const float4*)(z   + (size_t)row  * DDIM))[threadIdx.x];
  const float dx = zz.x - e.x, dy = zz.y - e.y, dz = zz.z - e.z, dw = zz.w - e.w;
  float4 o;   // bit-faithful to  z + stop_gradient(q - z)
  o.x = zz.x + (e.x - zz.x); o.y = zz.y + (e.y - zz.y);
  o.z = zz.z + (e.z - zz.z); o.w = zz.w + (e.w - zz.w);
  ((float4*)(out + (size_t)row * DDIM))[threadIdx.x] = o;

  float s = dx * dx + dy * dy + dz * dz + dw * dw;
#pragma unroll
  for (int m = 16; m >= 1; m >>= 1) s += __shfl_xor(s, m, 32);
  __shared__ float red[8];
  if ((threadIdx.x & 31) == 0) red[threadIdx.x >> 5] = s;
  __syncthreads();
  if (threadIdx.x == 0) {
    float t = 0.f;
#pragma unroll
    for (int i = 0; i < 8; ++i) t += red[i];
    partials[row] = t;
  }
}

// ---------------------------------------------------------------------------
// Kernel 5: loss = 1.25 * mean((z - q)^2), deterministic tree reduce.
// ---------------------------------------------------------------------------
__global__ __launch_bounds__(256) void vq_loss(const float* __restrict__ partials,
                                               float* __restrict__ out_loss) {
  float s = 0.f;
  for (int i = threadIdx.x; i < NROWS; i += 256) s += partials[i];
  __shared__ float red[256];
  red[threadIdx.x] = s;
  __syncthreads();
  for (int off = 128; off > 0; off >>= 1) {
    if (threadIdx.x < off) red[threadIdx.x] += red[threadIdx.x + off];
    __syncthreads();
  }
  if (threadIdx.x == 0)
    out_loss[0] = 1.25f * red[0] / ((float)NROWS * (float)DDIM);
}

// ---------------------------------------------------------------------------
extern "C" void kernel_launch(void* const* d_in, const int* in_sizes, int n_in,
                              void* d_out, int out_size, void* d_ws, size_t ws_size,
                              hipStream_t stream) {
  (void)in_sizes; (void)n_in; (void)out_size; (void)ws_size;
  const float* z   = (const float*)d_in[0];   // [8192,1024] f32
  const float* emb = (const float*)d_in[1];   // [8192,1024] f32
  float* out = (float*)d_out;                 // [8192*1024] quantized_st + [1] loss

  char* ws = (char*)d_ws;
  float* enorm    = (float*)ws;                                   // 8192 f
  float* partS    = (float*)(ws + 32768);                         // 8192*8 f
  int*   partI    = (int*)  (ws + 32768 + 262144);                // 8192*8 i
  int*   argmin   = (int*)  (ws + 32768 + 2 * 262144);            // 8192 i
  float* partials = (float*)(ws + 32768 + 2 * 262144 + 32768);    // 8192 f
  bf16_t* zh = (bf16_t*)(ws + 622592);                            // 16 MB each
  bf16_t* zl = zh + (size_t)NROWS * DDIM;
  bf16_t* ehp = zl + (size_t)NROWS * DDIM;
  bf16_t* elp = ehp + (size_t)KCODES * DDIM;

  vq_precvt<<<NROWS * DDIM / 1024, 256, 0, stream>>>(z, zh, zl);
  vq_precvt<<<KCODES * DDIM / 1024, 256, 0, stream>>>(emb, ehp, elp);
  vq_enorm<<<KCODES, 256, 0, stream>>>(emb, enorm);

  dim3 grid(NROWS / TILE, CSPLIT);
  vq_gemm_argmin<<<grid, 256, 0, stream>>>(zh, zl, ehp, elp, enorm, partS, partI);

  vq_merge<<<NROWS / 256, 256, 0, stream>>>(partS, partI, argmin);
  vq_quantize<<<NROWS, 256, 0, stream>>>(z, emb, argmin, out, partials);
  vq_loss<<<1, 256, 0, stream>>>(partials, out + (size_t)NROWS * DDIM);
}
